// WindowedCrossAttention_83013127897147
// MI455X (gfx1250) — matrix operand order repaired and checked by reading, not compiled, not run on hardware
//
#include <hip/hip_runtime.h>
#include <hip/hip_bf16.h>

typedef __attribute__((ext_vector_type(16))) __bf16        v16bf;
typedef __attribute__((ext_vector_type(2)))  __bf16        v2bf;
typedef __attribute__((ext_vector_type(8)))  float         v8f;
typedef __attribute__((ext_vector_type(8)))  unsigned int  v8u;

#define DIM   256
#define NTOK  64
#define HEADS 8
#define HD    32
#define QK_SCALE 0.17677669529663687f   // 1/sqrt(32)

#if defined(__has_builtin)
#if __has_builtin(__builtin_amdgcn_cvt_pk_bf16_f32)
#define HAVE_CVT_PK_BF16 1
#endif
#endif

__device__ __forceinline__ unsigned short f2bf(float f) {
#ifdef HAVE_CVT_PK_BF16
  v2bf r = __builtin_amdgcn_cvt_pk_bf16_f32(f, 0.0f);
  return (unsigned short)__builtin_bit_cast(unsigned, r);
#else
  unsigned u = __builtin_bit_cast(unsigned, f);
  u += 0x7FFFu + ((u >> 16) & 1u);          // round-to-nearest-even
  return (unsigned short)(u >> 16);
#endif
}
__device__ __forceinline__ unsigned pack2bf(float a, float b) {
#ifdef HAVE_CVT_PK_BF16
  v2bf r = __builtin_amdgcn_cvt_pk_bf16_f32(a, b);
  return __builtin_bit_cast(unsigned, r);
#else
  unsigned ua = __builtin_bit_cast(unsigned, a);
  ua += 0x7FFFu + ((ua >> 16) & 1u);
  unsigned ub = __builtin_bit_cast(unsigned, b);
  ub += 0x7FFFu + ((ub >> 16) & 1u);
  return (ua >> 16) | (ub & 0xFFFF0000u);
#endif
}

__device__ __forceinline__ v8f wmma_bf16(v16bf a, v16bf b, v8f c) {
  return __builtin_amdgcn_wmma_f32_16x16x32_bf16(
      /*neg_a=*/false, a, /*neg_b=*/false, b,
      /*c_mod=*/(short)0, c, /*reuse_a=*/false, /*reuse_b=*/false);
}

// A fragment: 16x32 bf16 tile, rows row0.., K-window [k0,k0+32), row-major src (ld elems).
// Contiguous-k order used here (matches load_b): lane holds row M=lane%16; K {0-15} (lanes 0-15) / {16-31} (lanes 16-31).
__device__ __forceinline__ v16bf load_a(const unsigned short* base, int ld,
                                        int row0, int k0, int lane) {
  const unsigned short* p = base + (row0 + (lane & 15)) * ld + k0;
  const int kb0 = (lane & 16) ? 16 : 0;
  const int kb1 = (lane & 16) ? 24 : 8;
  v8u u;
#pragma unroll
  for (int i = 0; i < 4; ++i) u[i]     = *(const unsigned*)(p + kb0 + 2 * i);
#pragma unroll
  for (int i = 0; i < 4; ++i) u[4 + i] = *(const unsigned*)(p + kb1 + 2 * i);
  return __builtin_bit_cast(v16bf, u);
}

// B fragment: 32x16 bf16 tile. B[k][n] = base[(n0+n)*ldk + k0 + k] (k-pairs contiguous).
// Contiguous-k order used here (matches load_a): lane holds col N=lane%16; K {0-15} (lanes 0-15) / {16-31} (lanes 16-31).
__device__ __forceinline__ v16bf load_b(const unsigned short* base, int ldk,
                                        int n0, int k0, int lane) {
  const unsigned short* p = base + (n0 + (lane & 15)) * ldk + k0 + ((lane & 16) ? 16 : 0);
  v8u u;
#pragma unroll
  for (int i = 0; i < 8; ++i) u[i] = *(const unsigned*)(p + 2 * i);
  return __builtin_bit_cast(v16bf, u);
}

// ---- per-launch prep: weight transpose/convert + rel-bias expansion into
// ---- C-fragment order: bias_exp[h][lane][mt][nt][g]  (f32, 128 KB, L2-resident)
__global__ void __launch_bounds__(256) prep_weights(
    const float* __restrict__ Wq, const float* __restrict__ Wkv,
    const float* __restrict__ Wp, const float* __restrict__ bias_table,
    unsigned short* __restrict__ WqT, unsigned short* __restrict__ WkvT,
    unsigned short* __restrict__ WpT, float* __restrict__ bias_exp) {
  int idx = blockIdx.x * 256 + threadIdx.x;
  if (idx < 65536) {                       // WqT[n][k] = Wq[k][n]
    int n = idx >> 8, k = idx & 255;
    WqT[n * 256 + k] = f2bf(Wq[k * 256 + n]);
  } else if (idx < 196608) {               // WkvT[n][k] = Wkv[k][n], n in [0,512)
    int e = idx - 65536;
    int n = e >> 8, k = e & 255;
    WkvT[n * 256 + k] = f2bf(Wkv[k * 512 + n]);
  } else if (idx < 262144) {               // WpT[n][k] = Wp[k][n]
    int e = idx - 196608;
    int n = e >> 8, k = e & 255;
    WpT[n * 256 + k] = f2bf(Wp[k * 256 + n]);
  } else if (idx < 294912) {               // bias fragment expansion
    int e    = idx - 262144;               // [h][lane][mt][nt][g]
    int h    = e >> 12;
    int rem  = e & 4095;
    int lane = rem >> 7;
    int c    = rem & 127;
    int mt   = c >> 5;
    int nt   = (c >> 3) & 3;
    int g    = c & 7;
    int M = mt * 16 + g + ((lane & 16) ? 8 : 0);
    int N = nt * 16 + (lane & 15);
    int my = M >> 3, mx = M & 7, ny = N >> 3, nx = N & 7;
    int tidx = (my - ny + 7) * 15 + (mx - nx + 7);
    bias_exp[e] = bias_table[tidx * HEADS + h];
  }
}

// ---- fused windowed cross-attention: 1 block = 1 window, 8 waves ----
__global__ void __launch_bounds__(256) wca_main(
    const float* __restrict__ qtok, const float* __restrict__ stok,
    const float* __restrict__ bq,   const float* __restrict__ bkv,
    const float* __restrict__ bp,
    const unsigned short* __restrict__ WqT,
    const unsigned short* __restrict__ WkvT,
    const unsigned short* __restrict__ WpT,
    const float* __restrict__ bias_exp,
    float* __restrict__ out) {
  extern __shared__ char smem_raw[];
  unsigned short* sm = (unsigned short*)smem_raw;
  unsigned short* Xq = sm;              // [64][256] bf16   (phase A)
  unsigned short* Xs = sm + 16384;      // [64][256] bf16   (phase A)
  unsigned short* Pm = sm;              // [8][64][64] bf16 (phase B, overlays Xq/Xs)
  unsigned short* Qs = sm + 32768;      // [64][256] bf16, scaled Q; later O overlay
  unsigned short* Ks = sm + 49152;      // [64][256] bf16
  unsigned short* Vt = sm + 65536;      // [256][64] bf16 (V transposed: [chan][token])
  unsigned short* Os = Qs;              // O overlays Q after barrier 2

  const int tid  = threadIdx.x;
  const int lane = tid & 31;
  const int wave = tid >> 5;
  const int b    = blockIdx.x;
  const int hi   = (lane >> 4) & 1;     // C-layout: rows g (+8 for high half-wave)
  const int ln   = lane & 15;           // C-layout: column within tile

  // -------- Phase A0: stage tokens to LDS as bf16 --------
  {
    const float4* q4 = (const float4*)(qtok + (size_t)b * NTOK * DIM);
    const float4* s4 = (const float4*)(stok + (size_t)b * NTOK * DIM);
    unsigned* xq = (unsigned*)Xq;
    unsigned* xs = (unsigned*)Xs;
    for (int i = tid; i < 4096; i += 256) {
      float4 v = q4[i];
      xq[2 * i]     = pack2bf(v.x, v.y);
      xq[2 * i + 1] = pack2bf(v.z, v.w);
      float4 w = s4[i];
      xs[2 * i]     = pack2bf(w.x, w.y);
      xs[2 * i + 1] = pack2bf(w.z, w.w);
    }
  }
  __syncthreads();

  // -------- Phase A1: projections; wave owns row-block mt, column range by half ----
  // A-fragments for the row block are loaded ONCE into registers, then reused
  // across all column tiles (inner loop = 2x global_load_b128 + 1 WMMA).
  const int mta  = wave & 3;            // row block (m-tile) this wave owns
  const int half = wave >> 2;           // column-range selector

  { // ---- Q projection: nt in [half*8, half*8+8) ----
    v16bf afr[8];
#pragma unroll
    for (int kk = 0; kk < 8; ++kk) afr[kk] = load_a(Xq, DIM, mta * 16, kk * 32, lane);
    for (int i = 0; i < 8; ++i) {
      const int nt = half * 8 + i;
      const int colbase = nt * 16;
      const int n = colbase + ln;
      const float bias = bq[n];
      v8f acc;
#pragma unroll
      for (int g = 0; g < 8; ++g) acc[g] = bias;
#pragma unroll
      for (int kk = 0; kk < 8; ++kk)
        acc = wmma_bf16(afr[kk], load_b(WqT, DIM, colbase, kk * 32, lane), acc);
#pragma unroll
      for (int g = 0; g < 8; ++g) {
        int row = mta * 16 + g + hi * 8;
        Qs[row * DIM + n] = f2bf(acc[g] * QK_SCALE);
      }
    }
  }

  { // ---- KV projection: nt in [half*16, half*16+16); half 0 -> K, half 1 -> V ----
    v16bf afr[8];
#pragma unroll
    for (int kk = 0; kk < 8; ++kk) afr[kk] = load_a(Xs, DIM, mta * 16, kk * 32, lane);
    for (int i = 0; i < 16; ++i) {
      const int nt = half * 16 + i;
      const int colbase = nt * 16;
      const int n = colbase + ln;
      const float bias = bkv[n];
      v8f acc;
#pragma unroll
      for (int g = 0; g < 8; ++g) acc[g] = bias;
#pragma unroll
      for (int kk = 0; kk < 8; ++kk)
        acc = wmma_bf16(afr[kk], load_b(WkvT, DIM, colbase, kk * 32, lane), acc);
      if (half == 0) {                  // K half of KV (wave-uniform branch)
#pragma unroll
        for (int g = 0; g < 8; ++g) {
          int row = mta * 16 + g + hi * 8;
          Ks[row * DIM + n] = f2bf(acc[g]);
        }
      } else {                          // V half -> store transposed
#pragma unroll
        for (int g = 0; g < 8; ++g) {
          int row = mta * 16 + g + hi * 8;
          Vt[(n - 256) * NTOK + row] = f2bf(acc[g]);
        }
      }
    }
  }
  __syncthreads();

  // -------- Phase B: per-head attention; head h = wave --------
  const int h = wave;
  v8f s[4][4];
  // seed accumulators with pre-expanded rel-pos bias (fragment-order, wide loads)
  {
    const float4* bf4 = (const float4*)(bias_exp + ((h * 32 + lane) << 7));
#pragma unroll
    for (int mt = 0; mt < 4; ++mt) {
#pragma unroll
      for (int nt = 0; nt < 4; ++nt) {
        float4 lo  = bf4[mt * 8 + nt * 2];
        float4 hi4 = bf4[mt * 8 + nt * 2 + 1];
        s[mt][nt][0] = lo.x;  s[mt][nt][1] = lo.y;
        s[mt][nt][2] = lo.z;  s[mt][nt][3] = lo.w;
        s[mt][nt][4] = hi4.x; s[mt][nt][5] = hi4.y;
        s[mt][nt][6] = hi4.z; s[mt][nt][7] = hi4.w;
      }
    }
  }
  // S = Q_h K_h^T + bias: hold all 4 K-fragments, stream A per m-tile
  {
    v16bf bfr[4];
#pragma unroll
    for (int nt = 0; nt < 4; ++nt) bfr[nt] = load_b(Ks, DIM, nt * 16, h * HD, lane);
#pragma unroll
    for (int mt = 0; mt < 4; ++mt) {
      v16bf a = load_a(Qs, DIM, mt * 16, h * HD, lane);
#pragma unroll
      for (int nt = 0; nt < 4; ++nt) s[mt][nt] = wmma_bf16(a, bfr[nt], s[mt][nt]);
    }
  }
  // row softmax in registers (rows live across 16 lanes of a half-wave)
  unsigned short* Ph = Pm + h * NTOK * NTOK;
#pragma unroll
  for (int mt = 0; mt < 4; ++mt) {
#pragma unroll
    for (int g = 0; g < 8; ++g) {
      float m0 = fmaxf(fmaxf(s[mt][0][g], s[mt][1][g]),
                       fmaxf(s[mt][2][g], s[mt][3][g]));
#pragma unroll
      for (int d = 1; d < 16; d <<= 1) m0 = fmaxf(m0, __shfl_xor(m0, d, 32));
      float e0 = __expf(s[mt][0][g] - m0);
      float e1 = __expf(s[mt][1][g] - m0);
      float e2 = __expf(s[mt][2][g] - m0);
      float e3 = __expf(s[mt][3][g] - m0);
      float sum = e0 + e1 + e2 + e3;
#pragma unroll
      for (int d = 1; d < 16; d <<= 1) sum += __shfl_xor(sum, d, 32);
      float r = 1.0f / sum;
      int row = mt * 16 + g + hi * 8;
      Ph[row * NTOK +  0 + ln] = f2bf(e0 * r);
      Ph[row * NTOK + 16 + ln] = f2bf(e1 * r);
      Ph[row * NTOK + 32 + ln] = f2bf(e2 * r);
      Ph[row * NTOK + 48 + ln] = f2bf(e3 * r);
    }
  }
  __syncthreads();   // all Q/K reads finished -> safe to overlay O onto Qs

  // O_h = P_h V_h: hold all 4 V-fragments, stream P per m-tile
  {
    v16bf vb[4];
#pragma unroll
    for (int nt = 0; nt < 2; ++nt) {
      vb[nt * 2]     = load_b(Vt, NTOK, h * HD + nt * 16, 0, lane);
      vb[nt * 2 + 1] = load_b(Vt, NTOK, h * HD + nt * 16, 32, lane);
    }
#pragma unroll
    for (int mt = 0; mt < 4; ++mt) {
      v16bf a0 = load_a(Ph, NTOK, mt * 16, 0, lane);
      v16bf a1 = load_a(Ph, NTOK, mt * 16, 32, lane);
#pragma unroll
      for (int nt = 0; nt < 2; ++nt) {
        v8f acc;
#pragma unroll
        for (int g = 0; g < 8; ++g) acc[g] = 0.0f;
        acc = wmma_bf16(a0, vb[nt * 2], acc);
        acc = wmma_bf16(a1, vb[nt * 2 + 1], acc);
#pragma unroll
        for (int g = 0; g < 8; ++g) {
          int row = mt * 16 + g + hi * 8;
          Os[row * DIM + h * HD + nt * 16 + ln] = f2bf(acc[g]);
        }
      }
    }
  }
  __syncthreads();

  // -------- Phase C: output projection (A-fragments cached), f32 stores --------
  {
    v16bf afr[8];
#pragma unroll
    for (int kk = 0; kk < 8; ++kk) afr[kk] = load_a(Os, DIM, mta * 16, kk * 32, lane);
    float* outb = out + (size_t)b * NTOK * DIM;
    for (int i = 0; i < 8; ++i) {
      const int nt = half * 8 + i;
      const int colbase = nt * 16;
      const int n = colbase + ln;
      const float bias = bp[n];
      v8f acc;
#pragma unroll
      for (int g = 0; g < 8; ++g) acc[g] = bias;
#pragma unroll
      for (int kk = 0; kk < 8; ++kk)
        acc = wmma_bf16(afr[kk], load_b(WpT, DIM, colbase, kk * 32, lane), acc);
#pragma unroll
      for (int g = 0; g < 8; ++g) {
        int row = mta * 16 + g + hi * 8;
        outb[row * DIM + n] = acc[g];
      }
    }
  }
}

extern "C" void kernel_launch(void* const* d_in, const int* in_sizes, int n_in,
                              void* d_out, int out_size, void* d_ws, size_t ws_size,
                              hipStream_t stream) {
  const float* qtok       = (const float*)d_in[0];
  const float* stok       = (const float*)d_in[1];
  const float* Wq         = (const float*)d_in[2];
  const float* bq         = (const float*)d_in[3];
  const float* Wkv        = (const float*)d_in[4];
  const float* bkv        = (const float*)d_in[5];
  const float* bias_table = (const float*)d_in[6];
  const float* Wp         = (const float*)d_in[7];
  const float* bp         = (const float*)d_in[8];
  float* out = (float*)d_out;

  unsigned short* WqT  = (unsigned short*)d_ws;            // 256*256 bf16 (128 KB)
  unsigned short* WkvT = WqT + 256 * 256;                  // 512*256 bf16 (256 KB)
  unsigned short* WpT  = WkvT + 512 * 256;                 // 256*256 bf16 (128 KB)
  float* bias_exp = (float*)((char*)d_ws + 524288);        // [8][32][128] f32 (128 KB)

  const int B = in_sizes[0] / (NTOK * DIM);

  prep_weights<<<1152, 256, 0, stream>>>(Wq, Wkv, Wp, bias_table,
                                         WqT, WkvT, WpT, bias_exp);

  const int lds_bytes = 160 * 1024;
  hipFuncSetAttribute((const void*)wca_main,
                      hipFuncAttributeMaxDynamicSharedMemorySize, lds_bytes);
  wca_main<<<B, 256, lds_bytes, stream>>>(qtok, stok, bq, bkv, bp,
                                          WqT, WkvT, WpT, bias_exp, out);
}
